// Pointer_73229192396864
// MI455X (gfx1250) — matrix operand order repaired and checked by reading, not compiled, not run on hardware
//
#include <hip/hip_runtime.h>

// MI455X / gfx1250, wave32. Pipeline:
//  1) s[b,q]   = tanh(u@Wu^T + b) . vb        (WMMA bf16 GEMM, fused epilogue)
//  2) a        = softmax_q(s); r = a @ u      (chunked atomic reduction)
//  3) proj     = h @ Wh^T  -> bf16 scratch    (WMMA bf16 GEMM, stored once, used twice)
//  4) y1=r@Wha^T; s2[b,p]=sum_h tanh(proj+y1)*vb; p1=softmax_p
//  5) c = p1 @ h; r2 = GRU(c, r)
//  6) y2=r2@Wha^T; s3; p2=softmax_p
//
// GEMM blocking: 256 thr = 8 waves per block, 128 rows. Wave tile = 16 rows x
// 64 cols (4 f32 accumulators), K staged in 64-wide slabs (f32->bf16 in LDS).
// 8 WMMAs per barrier pair; A slab re-read from L2 only 8x per block.

typedef __attribute__((ext_vector_type(16))) __bf16 v16bf;
typedef __attribute__((ext_vector_type(8)))  float  v8f;
typedef __attribute__((ext_vector_type(8)))  __bf16 bf16x8;

#define DD 512
#define BB 32
#define LQQ 4096
#define LPP 4096

__device__ __forceinline__ float red_add16(float v) {
  v += __shfl_xor(v, 1, 32);
  v += __shfl_xor(v, 2, 32);
  v += __shfl_xor(v, 4, 32);
  v += __shfl_xor(v, 8, 32);
  return v;
}

// stage `nf4` float4s from src into dst (f32 -> bf16)
__device__ __forceinline__ void stage_bf16(const float* __restrict__ src,
                                           __bf16* __restrict__ dst, int nf4) {
#pragma unroll
  for (int i = 0; i < 8; ++i) {
    if (i >= nf4) break;
    float4 f = ((const float4*)src)[i];
    dst[4*i + 0] = (__bf16)f.x;
    dst[4*i + 1] = (__bf16)f.y;
    dst[4*i + 2] = (__bf16)f.z;
    dst[4*i + 3] = (__bf16)f.w;
  }
}

// Build A fragment (contiguous k = 16*hi + s, matching frag_b) from an LDS row.
__device__ __forceinline__ v16bf frag_a(const __bf16* __restrict__ rowp, int ka) {
  v16bf av;
#pragma unroll
  for (int j = 0; j < 8; ++j) av[j] = rowp[ka + j];
#pragma unroll
  for (int j = 0; j < 8; ++j) av[8 + j] = rowp[8 + ka + j];
  return av;
}

// Build B fragment (contiguous k = 16*hi + j, matching frag_a) from an LDS row (column of W).
__device__ __forceinline__ v16bf frag_b(const __bf16* __restrict__ rowp, int kb) {
  v16bf bv;
#pragma unroll
  for (int j = 0; j < 16; ++j) bv[j] = rowp[kb + j];
  return bv;
}

// ---------------------------------------------------------------------------
// GEMM 1: s[row] = sum_h tanh( sum_d U[row,d]*W[h,d] + bias[h] ) * vb[b,h]
// ---------------------------------------------------------------------------
__global__ __launch_bounds__(256)
void gemm_tanh_dot_kernel(const float* __restrict__ U, const float* __restrict__ W,
                          const float* __restrict__ bias, const float* __restrict__ vb,
                          float* __restrict__ sOut)
{
  __shared__ __bf16 lds_a[128][64];   // 16 KB
  __shared__ __bf16 lds_b[64][64];    //  8 KB
  const int tid  = threadIdx.x;
  const int wave = tid >> 5;
  const int lane = tid & 31;
  const int lo   = lane & 15;
  const int hi   = lane >> 4;
  const long row0 = (long)blockIdx.x * 128;
  const int  bidx = (int)(row0 / LQQ);     // 128 | LQ, so one b per block

  const int ar = tid >> 1;                 // A stage: 128 rows x 64 cols
  const int ac = (tid & 1) * 32;
  const int br = tid >> 2;                 // B stage: 64 rows x 64 cols
  const int bc = (tid & 3) * 16;

  float sacc[8];
#pragma unroll
  for (int i = 0; i < 8; ++i) sacc[i] = 0.f;

  for (int h0 = 0; h0 < DD; h0 += 64) {
    v8f acc[4];
#pragma unroll
    for (int t = 0; t < 4; ++t) acc[t] = (v8f){};
    for (int k0 = 0; k0 < DD; k0 += 64) {
      __syncthreads();
      {
        const float* srcA = U + (row0 + ar) * DD + k0 + ac;
        if (k0 + 64 < DD) __builtin_prefetch(srcA + 64, 0, 3);
        stage_bf16(srcA, &lds_a[ar][ac], 8);
        const float* srcB = W + (h0 + br) * DD + k0 + bc;
        stage_bf16(srcB, &lds_b[br][bc], 4);
      }
      __syncthreads();
      const __bf16* arow = &lds_a[wave * 16 + lo][0];
      const int ka = hi * 16;
      const int kb = hi * 16;
#pragma unroll
      for (int s = 0; s < 2; ++s) {        // two 32-deep K sub-slabs
        v16bf av = frag_a(arow + 32 * s, ka);
#pragma unroll
        for (int t = 0; t < 4; ++t) {      // four 16-wide col tiles
          v16bf bv = frag_b(&lds_b[t * 16 + lo][32 * s], kb);
          acc[t] = __builtin_amdgcn_wmma_f32_16x16x32_bf16(
              false, av, false, bv, (short)0, acc[t], false, false);
        }
      }
    }
    // epilogue: bias + tanh + dot with vb, reduce over the 16 N-lanes
#pragma unroll
    for (int t = 0; t < 4; ++t) {
      const int hcol = h0 + t * 16 + lo;
      const float bn = bias[hcol];
      const float vn = vb[bidx * DD + hcol];
#pragma unroll
      for (int rI = 0; rI < 8; ++rI) {
        float val = tanhf(acc[t][rI] + bn) * vn;
        sacc[rI] += red_add16(val);
      }
    }
  }
  if (lo == 0) {
#pragma unroll
    for (int rI = 0; rI < 8; ++rI)
      sOut[row0 + wave * 16 + hi * 8 + rI] = sacc[rI];
  }
}

// ---------------------------------------------------------------------------
// GEMM 2: proj[row,h] = sum_d H[row,d]*W[h,d], stored bf16
// ---------------------------------------------------------------------------
__global__ __launch_bounds__(256)
void gemm_store_kernel(const float* __restrict__ H, const float* __restrict__ W,
                       __bf16* __restrict__ proj)
{
  __shared__ __bf16 lds_a[128][64];
  __shared__ __bf16 lds_b[64][64];
  const int tid  = threadIdx.x;
  const int wave = tid >> 5;
  const int lane = tid & 31;
  const int lo   = lane & 15;
  const int hi   = lane >> 4;
  const long row0 = (long)blockIdx.x * 128;
  const int ar = tid >> 1;
  const int ac = (tid & 1) * 32;
  const int br = tid >> 2;
  const int bc = (tid & 3) * 16;

  for (int h0 = 0; h0 < DD; h0 += 64) {
    v8f acc[4];
#pragma unroll
    for (int t = 0; t < 4; ++t) acc[t] = (v8f){};
    for (int k0 = 0; k0 < DD; k0 += 64) {
      __syncthreads();
      {
        const float* srcA = H + (row0 + ar) * DD + k0 + ac;
        if (k0 + 64 < DD) __builtin_prefetch(srcA + 64, 0, 3);
        stage_bf16(srcA, &lds_a[ar][ac], 8);
        const float* srcB = W + (h0 + br) * DD + k0 + bc;
        stage_bf16(srcB, &lds_b[br][bc], 4);
      }
      __syncthreads();
      const __bf16* arow = &lds_a[wave * 16 + lo][0];
      const int ka = hi * 16;
      const int kb = hi * 16;
#pragma unroll
      for (int s = 0; s < 2; ++s) {
        v16bf av = frag_a(arow + 32 * s, ka);
#pragma unroll
        for (int t = 0; t < 4; ++t) {
          v16bf bv = frag_b(&lds_b[t * 16 + lo][32 * s], kb);
          acc[t] = __builtin_amdgcn_wmma_f32_16x16x32_bf16(
              false, av, false, bv, (short)0, acc[t], false, false);
        }
      }
    }
#pragma unroll
    for (int t = 0; t < 4; ++t) {
      const int col = h0 + t * 16 + lo;
#pragma unroll
      for (int rI = 0; rI < 8; ++rI) {
        const long row = row0 + wave * 16 + hi * 8 + rI;
        proj[row * DD + col] = (__bf16)acc[t][rI];
      }
    }
  }
}

// ---------------------------------------------------------------------------
// s2[b*LP + p] = sum_h tanh(proj[(p*B+b), h] + y[b,h]) * vb[b,h]  (1 wave/row)
// ---------------------------------------------------------------------------
__global__ __launch_bounds__(256)
void attn_score_kernel(const __bf16* __restrict__ proj, const float* __restrict__ y,
                       const float* __restrict__ vb, float* __restrict__ sOut)
{
  const int lane = threadIdx.x & 31;
  const long g = (long)blockIdx.x * 8 + (threadIdx.x >> 5);
  const int  b = (int)(g % BB);
  const long p = g / BB;
  const int  h = lane * 16;
  const bf16x8* pp = (const bf16x8*)(proj + g * DD + h);
  bf16x8 v0 = pp[0];
  bf16x8 v1 = pp[1];
  const float* yr = y  + b * DD + h;
  const float* vr = vb + b * DD + h;
  float acc = 0.f;
#pragma unroll
  for (int i = 0; i < 8; ++i) {
    acc += tanhf((float)v0[i] + yr[i])     * vr[i];
    acc += tanhf((float)v1[i] + yr[8 + i]) * vr[8 + i];
  }
#pragma unroll
  for (int m = 16; m >= 1; m >>= 1) acc += __shfl_xor(acc, m, 32);
  if (lane == 0) sOut[(long)b * LPP + p] = acc;
}

// ---------------------------------------------------------------------------
// row-wise softmax: out[row, 0..C) = softmax(in[row, 0..C))
// ---------------------------------------------------------------------------
__global__ __launch_bounds__(256)
void softmax_rows_kernel(const float* __restrict__ in, float* __restrict__ out, int C)
{
  __shared__ float red[256];
  const int tid = threadIdx.x;
  const long row = blockIdx.x;
  const float* x = in + row * C;
  float m = -1e30f;
  for (int c = tid; c < C; c += 256) m = fmaxf(m, x[c]);
  red[tid] = m; __syncthreads();
  for (int s = 128; s > 0; s >>= 1) {
    if (tid < s) red[tid] = fmaxf(red[tid], red[tid + s]);
    __syncthreads();
  }
  m = red[0]; __syncthreads();
  float sum = 0.f;
  for (int c = tid; c < C; c += 256) sum += expf(x[c] - m);
  red[tid] = sum; __syncthreads();
  for (int s = 128; s > 0; s >>= 1) {
    if (tid < s) red[tid] += red[tid + s];
    __syncthreads();
  }
  const float inv = 1.f / red[0];
  for (int c = tid; c < C; c += 256) out[row * C + c] = expf(x[c] - m) * inv;
}

// ---------------------------------------------------------------------------
// out[b,d] += sum_{n in chunk} wgt[b*N+n] * X[b*bTerm + n*rowStride + d]
// ---------------------------------------------------------------------------
__global__ __launch_bounds__(256)
void weighted_sum_kernel(const float* __restrict__ wgt, const float* __restrict__ X,
                         float* __restrict__ out, int N, int nchunks,
                         long bTerm, long rowStride)
{
  const int b  = blockIdx.x / nchunks;
  const int ch = blockIdx.x % nchunks;
  const int chunk = N / nchunks;
  const int d = threadIdx.x;
  float a0 = 0.f, a1 = 0.f;
  const float* base = X + (long)b * bTerm;
  const int n0 = ch * chunk;
  for (int n = n0; n < n0 + chunk; ++n) {
    const float wv = wgt[(long)b * N + n];
    const float* rowp = base + (long)n * rowStride;
    a0 += wv * rowp[d];
    a1 += wv * rowp[d + 256];
  }
  atomicAdd(&out[b * DD + d], a0);
  atomicAdd(&out[b * DD + d + 256], a1);
}

// ---------------------------------------------------------------------------
// y[b,h] = dot(rvec[b,:], Wha[h,:])  -- tiny (32x512)x(512x512)
// ---------------------------------------------------------------------------
__global__ __launch_bounds__(256)
void matvec_kernel(const float* __restrict__ rvec, const float* __restrict__ Wha,
                   float* __restrict__ y)
{
  __shared__ float rl[DD];
  const long id = (long)blockIdx.x * 256 + threadIdx.x;
  const int b = (int)(id >> 9);
  const int h = (int)(id & 511);
  rl[threadIdx.x]       = rvec[b * DD + threadIdx.x];
  rl[threadIdx.x + 256] = rvec[b * DD + threadIdx.x + 256];
  __syncthreads();
  float acc = 0.f;
  const float* wr = Wha + (long)h * DD;
  for (int d = 0; d < DD; ++d) acc += rl[d] * wr[d];
  y[id] = acc;
}

// ---------------------------------------------------------------------------
// GRU cell
// ---------------------------------------------------------------------------
__device__ __forceinline__ float sigmoidf(float x) { return 1.f / (1.f + expf(-x)); }

__global__ __launch_bounds__(512)
void gru_kernel(const float* __restrict__ c, const float* __restrict__ r,
                const float* __restrict__ Wih, const float* __restrict__ Whh,
                const float* __restrict__ bih, const float* __restrict__ bhh,
                float* __restrict__ r2)
{
  __shared__ float cl[DD], rl[DD];
  const int b = blockIdx.x;
  const int j = threadIdx.x;
  cl[j] = c[b * DD + j];
  rl[j] = r[b * DD + j];
  __syncthreads();
  float gi[3], gh[3];
#pragma unroll
  for (int t = 0; t < 3; ++t) {
    const int row = t * DD + j;
    float a = bih[row], bb = bhh[row];
    const float* wi = Wih + (long)row * DD;
    const float* wh = Whh + (long)row * DD;
    for (int d = 0; d < DD; ++d) { a += cl[d] * wi[d]; bb += rl[d] * wh[d]; }
    gi[t] = a; gh[t] = bb;
  }
  const float rr = sigmoidf(gi[0] + gh[0]);
  const float zz = sigmoidf(gi[1] + gh[1]);
  const float nn = tanhf(gi[2] + rr * gh[2]);
  r2[b * DD + j] = (1.f - zz) * nn + zz * rl[j];
}

__global__ void zero_f32_kernel(float* __restrict__ p, int n)
{
  int i = blockIdx.x * 256 + threadIdx.x;
  if (i < n) p[i] = 0.f;
}

// ---------------------------------------------------------------------------
extern "C" void kernel_launch(void* const* d_in, const int* in_sizes, int n_in,
                              void* d_out, int out_size, void* d_ws, size_t ws_size,
                              hipStream_t stream)
{
  const float* h     = (const float*)d_in[0];
  const float* u     = (const float*)d_in[1];
  const float* v     = (const float*)d_in[2];   // (B, D, 1) -> vb[b,d] = v[b*D+d]
  const float* Wu_w  = (const float*)d_in[3];
  const float* Wu_b  = (const float*)d_in[4];
  const float* Wh_w  = (const float*)d_in[5];
  const float* Wha_w = (const float*)d_in[6];
  const float* W_ih  = (const float*)d_in[7];
  const float* W_hh  = (const float*)d_in[8];
  const float* b_ih  = (const float*)d_in[9];
  const float* b_hh  = (const float*)d_in[10];
  float* out = (float*)d_out;                   // p1 then p2, each B*LP

  char* w = (char*)d_ws;
  size_t off = 0;
  float*  sbuf  = (float*)(w + off); off += (size_t)BB * LQQ * sizeof(float);
  __bf16* proj  = (__bf16*)(w + off); off += (size_t)LPP * BB * DD * sizeof(__bf16);
  float*  rbuf  = (float*)(w + off); off += (size_t)BB * DD * sizeof(float);
  float*  ybuf  = (float*)(w + off); off += (size_t)BB * DD * sizeof(float);
  float*  cbuf  = (float*)(w + off); off += (size_t)BB * DD * sizeof(float);
  float*  r2buf = (float*)(w + off); off += (size_t)BB * DD * sizeof(float);
  float*  s2buf = (float*)(w + off); off += (size_t)BB * LPP * sizeof(float);

  // zero atomic accumulators
  zero_f32_kernel<<<(BB * DD + 255) / 256, 256, 0, stream>>>(rbuf, BB * DD);
  zero_f32_kernel<<<(BB * DD + 255) / 256, 256, 0, stream>>>(cbuf, BB * DD);

  // 1) scores s[b,q]
  gemm_tanh_dot_kernel<<<(BB * LQQ) / 128, 256, 0, stream>>>(u, Wu_w, Wu_b, v, sbuf);
  // 2) a = softmax_q(s) in place; r = a @ u
  softmax_rows_kernel<<<BB, 256, 0, stream>>>(sbuf, sbuf, LQQ);
  weighted_sum_kernel<<<BB * 8, 256, 0, stream>>>(sbuf, u, rbuf, LQQ, 8,
                                                  (long)LQQ * DD, (long)DD);
  // 3) proj = h @ Wh^T (bf16)
  gemm_store_kernel<<<(LPP * BB) / 128, 256, 0, stream>>>(h, Wh_w, proj);
  // 4) p1 = h_attn(r)
  matvec_kernel<<<(BB * DD) / 256, 256, 0, stream>>>(rbuf, Wha_w, ybuf);
  attn_score_kernel<<<(LPP * BB) / 8, 256, 0, stream>>>(proj, ybuf, v, s2buf);
  softmax_rows_kernel<<<BB, 256, 0, stream>>>(s2buf, out, LPP);
  // 5) c = p1 @ h ; r2 = GRU(c, r)
  weighted_sum_kernel<<<BB * 8, 256, 0, stream>>>(out, h, cbuf, LPP, 8,
                                                  (long)DD, (long)BB * DD);
  gru_kernel<<<BB, 512, 0, stream>>>(cbuf, rbuf, W_ih, W_hh, b_ih, b_hh, r2buf);
  // 6) p2 = h_attn(r2)
  matvec_kernel<<<(BB * DD) / 256, 256, 0, stream>>>(r2buf, Wha_w, ybuf);
  attn_score_kernel<<<(LPP * BB) / 8, 256, 0, stream>>>(proj, ybuf, v, s2buf);
  softmax_rows_kernel<<<BB, 256, 0, stream>>>(s2buf, out + (size_t)BB * LPP, LPP);

  (void)in_sizes; (void)n_in; (void)out_size; (void)ws_size;
}
